// D2LDotProductAttention_81389630259486
// MI455X (gfx1250) — matrix operand order repaired and checked by reading, not compiled, not run on hardware
//
#include <hip/hip_runtime.h>
#include <hip/hip_bf16.h>

typedef __attribute__((ext_vector_type(16))) _Float16 v16h;
typedef __attribute__((ext_vector_type(8)))  float    v8f;

#define B_  32
#define S_  2048
#define D_  64
#define TK  64            // key tile
#define WAVES 8
#define BLOCK_Q (WAVES * 16)
#define KSTRIDE 36        // padded LDS row stride (dwords): 16B-aligned, bank-spread

union Frag {
    v16h  h;
    uint4 q[2];
    unsigned int u[8];
};

static __device__ __forceinline__ unsigned int pk2h(float a, float b) {
    union { _Float16 h[2]; unsigned int u; } t;
    t.h[0] = (_Float16)a;
    t.h[1] = (_Float16)b;
    return t.u;
}

static __device__ __forceinline__ unsigned short h16(float a) {
    union { _Float16 h; unsigned short u; } t;
    t.h = (_Float16)a;
    return t.u;
}

static __device__ __forceinline__ v8f wmma_f16(v16h a, v16h b, v8f c) {
    // D = A(16x32 f16) * B(32x16 f16) + C(16x16 f32)
    return __builtin_amdgcn_wmma_f32_16x16x32_f16(
        false, a, false, b, (short)0, c, false, false);
}

static __device__ __forceinline__ float rmax16(float v) {
#pragma unroll
    for (int o = 1; o < 16; o <<= 1)
        v = fmaxf(v, __shfl_xor(v, o, 32));
    return v;
}

// Workgroup barrier that publishes only LDS (DScnt), NOT outstanding global
// loads: __syncthreads()'s full fence would drain loadcnt and stall the
// register-staged prefetch. DS stores are drained, then split-barrier.
static __device__ __forceinline__ void barrier_lds_only() {
    asm volatile("s_wait_dscnt 0x0" ::: "memory");
    __builtin_amdgcn_s_barrier();
}

__global__ __launch_bounds__(256, 1)
void flash_attn_kernel(const float* __restrict__ Q,
                       const float* __restrict__ K,
                       const float* __restrict__ V,
                       const int*   __restrict__ valid_lens,
                       float*       __restrict__ O) {
    // Double-buffered tiles:
    // sK[buf][col][d2]: half2 of (K[col][2*d2], K[col][2*d2+1]); col-major so each
    //   B-fragment (d2 base..base+3 and base+8..base+11 per lane) is two ds_load_b128.
    __shared__ __align__(16) unsigned int sK[2][TK * KSTRIDE];        // 18 KB
    // sV[buf][d][k2]: half2 of (V[2*k2][d], V[2*k2+1][d]).
    __shared__ __align__(16) unsigned int sV[2][D_ * KSTRIDE];        // 18 KB
    // per-wave P scratch: row-major [16][TK] halves, padded to 2*KSTRIDE halves/row
    __shared__ __align__(16) unsigned int sP[WAVES][16 * KSTRIDE];    // 18 KB

    const int tid  = threadIdx.x;
    const int wave = tid >> 5;
    const int lane = tid & 31;
    const int n    = lane & 15;   // column index inside a 16-wide tile / row for A
    const int hh   = lane >> 4;   // half-wave select

    const int blocks_per_batch = S_ / BLOCK_Q;
    const int b    = blockIdx.x / blocks_per_batch;
    const int qblk = blockIdx.x % blocks_per_batch;
    const int q0   = qblk * BLOCK_Q + wave * 16;

    const float* Qb = Q + (size_t)b * S_ * D_;
    const float* Kb = K + (size_t)b * S_ * D_;
    const float* Vb = V + (size_t)b * S_ * D_;
    float*       Ob = O + (size_t)b * S_ * D_;

    // 1/sqrt(64) * log2(e): scores land in log2 domain -> exp2f for softmax
    const float qscale = 0.125f * 1.4426950408889634f;

    // ---- Load Q fragments once (A-layout, 16x32 f16, two K-steps for D=64) ----
    Frag qf[2];
#pragma unroll
    for (int step = 0; step < 2; ++step) {
        const float* qrow = Qb + (size_t)(q0 + n) * D_ + step * 32 + 8 * hh;
        const float4 f0 = *(const float4*)(qrow + 0);
        const float4 f1 = *(const float4*)(qrow + 4);
        const float4 f2 = *(const float4*)(qrow + 16);
        const float4 f3 = *(const float4*)(qrow + 20);
        qf[step].u[0] = pk2h(f0.x * qscale, f0.y * qscale);
        qf[step].u[1] = pk2h(f0.z * qscale, f0.w * qscale);
        qf[step].u[2] = pk2h(f1.x * qscale, f1.y * qscale);
        qf[step].u[3] = pk2h(f1.z * qscale, f1.w * qscale);
        qf[step].u[4] = pk2h(f2.x * qscale, f2.y * qscale);
        qf[step].u[5] = pk2h(f2.z * qscale, f2.w * qscale);
        qf[step].u[6] = pk2h(f3.x * qscale, f3.y * qscale);
        qf[step].u[7] = pk2h(f3.z * qscale, f3.w * qscale);
    }

    // all-ones B fragment (f16 1.0 everywhere): row sums l = P * ones via WMMA
    Frag ones;
#pragma unroll
    for (int j = 0; j < 8; ++j) ones.u[j] = 0x3C003C00u;

    v8f acc[4];
#pragma unroll
    for (int dc = 0; dc < 4; ++dc) acc[dc] = (v8f)(0.0f);
    v8f acc_l = (v8f)(0.0f);      // running softmax denominator (C-layout replicated)

    float m_run[8];
#pragma unroll
    for (int j = 0; j < 8; ++j) m_run[j] = -1e30f;

    const int valid = valid_lens[b];            // uniform across block
    const int nT    = (valid + TK - 1) / TK;    // total tiles (>= 1)

    // ---- Register staging for software pipeline (global -> regs) ----
    float2 kreg[8], vreg[8];

    auto load_tile = [&](int kbase) {
#pragma unroll
        for (int r = 0; r < 8; ++r) {
            const int i   = tid + r * 256;
            const int col = i >> 5;      // 0..63 key within tile
            const int d2  = i & 31;      // 0..31
            kreg[r] = *(const float2*)(Kb + (size_t)(kbase + col) * D_ + 2 * d2);
        }
#pragma unroll
        for (int r = 0; r < 8; ++r) {
            const int i  = tid + r * 256;
            const int k2 = i >> 6;       // 0..31
            const int d  = i & 63;       // 0..63 (consecutive tid -> coalesced)
            const float* p = Vb + (size_t)(kbase + 2 * k2) * D_ + d;
            vreg[r].x = p[0];
            vreg[r].y = p[D_];
        }
    };

    auto store_tile = [&](int buf) {
#pragma unroll
        for (int r = 0; r < 8; ++r) {
            const int i   = tid + r * 256;
            const int col = i >> 5;
            const int d2  = i & 31;
            sK[buf][col * KSTRIDE + d2] = pk2h(kreg[r].x, kreg[r].y);
        }
#pragma unroll
        for (int r = 0; r < 8; ++r) {
            const int i  = tid + r * 256;
            const int k2 = i >> 6;
            const int d  = i & 63;
            sV[buf][d * KSTRIDE + k2] = pk2h(vreg[r].x, vreg[r].y);
        }
    };

    auto compute_tile = [&](int kbase, int buf, bool masked) {
        const unsigned int* sKb = &sK[buf][0];
        const unsigned int* sVb = &sV[buf][0];

        // ---- Scores: 4 sub-tiles of 16 keys ----
        v8f s[4];
#pragma unroll
        for (int t = 0; t < 4; ++t) {
            v8f c = (v8f)(0.0f);
#pragma unroll
            for (int step = 0; step < 2; ++step) {
                // B-layout: lanes 0-15 hold K=0..7,16..23 (pairs), 16-31 hold K=8..15,24..31
                const unsigned int* kb =
                    &sKb[(t * 16 + n) * KSTRIDE + step * 16 + 4 * hh];
                Frag kf;
                kf.q[0] = *(const uint4*)(kb + 0);
                kf.q[1] = *(const uint4*)(kb + 8);
                c = wmma_f16(qf[step].h, kf.h, c);
            }
            s[t] = c;
        }

        // ---- Mask (boundary tile only) + online softmax (log2 domain) ----
        unsigned short* pbase = (unsigned short*)&sP[wave][0];
#pragma unroll
        for (int j = 0; j < 8; ++j) {
            float mx = m_run[j];
#pragma unroll
            for (int t = 0; t < 4; ++t) {
                if (masked) {
                    const int key = kbase + t * 16 + n;
                    if (key >= valid) s[t][j] = -1e30f;
                }
                mx = fmaxf(mx, s[t][j]);
            }
            mx = rmax16(mx);

            const float corr = exp2f(m_run[j] - mx);
            m_run[j] = mx;
            acc_l[j] *= corr;
#pragma unroll
            for (int dc = 0; dc < 4; ++dc) acc[dc][j] *= corr;

            const int row = j + 8 * hh;
#pragma unroll
            for (int t = 0; t < 4; ++t) {
                const float p = exp2f(s[t][j] - mx);
                pbase[row * (2 * KSTRIDE) + t * 16 + n] = h16(p);
            }
        }

        // ---- P * V and row sums P * 1 (DS in-order per wave) ----
#pragma unroll
        for (int kstep = 0; kstep < 2; ++kstep) {
            // A-layout pair indices: {base..base+3} and {base+8..base+11}
            const unsigned int* pb =
                &sP[wave][n * KSTRIDE + 16 * kstep + 4 * hh];
            Frag pf;
            pf.q[0] = *(const uint4*)(pb + 0);
            pf.q[1] = *(const uint4*)(pb + 8);

            acc_l = wmma_f16(pf.h, ones.h, acc_l);   // denominator on the matrix unit
#pragma unroll
            for (int dc = 0; dc < 4; ++dc) {
                const unsigned int* vb =
                    &sVb[(dc * 16 + n) * KSTRIDE + 16 * kstep + 4 * hh];
                Frag vf;
                vf.q[0] = *(const uint4*)(vb + 0);
                vf.q[1] = *(const uint4*)(vb + 8);
                acc[dc] = wmma_f16(pf.h, vf.h, acc[dc]);
            }
        }
    };

    // ---- Software-pipelined main loop (double-buffered LDS) ----
    load_tile(0);
    store_tile(0);
    for (int kt = 0; kt < nT - 1; ++kt) {
        load_tile((kt + 1) * TK);            // global loads stay outstanding
        barrier_lds_only();                  //  across the barrier (DScnt only)
        compute_tile(kt * TK, kt & 1, false);
        store_tile((kt + 1) & 1);            // waits loadcnt here, after compute
    }
    barrier_lds_only();
    // Final tile: always masked body (select is a no-op when rem == 0).
    compute_tile((nT - 1) * TK, (nT - 1) & 1, true);

    // ---- Epilogue: normalize and store ----
#pragma unroll
    for (int j = 0; j < 8; ++j) {
        const float inv = 1.0f / acc_l[j];
        const int row = q0 + j + 8 * hh;
#pragma unroll
        for (int dc = 0; dc < 4; ++dc) {
            Ob[(size_t)row * D_ + dc * 16 + n] = acc[dc][j] * inv;
        }
    }
}

extern "C" void kernel_launch(void* const* d_in, const int* in_sizes, int n_in,
                              void* d_out, int out_size, void* d_ws, size_t ws_size,
                              hipStream_t stream) {
    const float* Q = (const float*)d_in[0];
    const float* K = (const float*)d_in[1];
    const float* V = (const float*)d_in[2];
    const int* valid_lens = (const int*)d_in[3];
    float* O = (float*)d_out;

    dim3 grid(B_ * (S_ / BLOCK_Q));
    dim3 block(256);
    flash_attn_kernel<<<grid, block, 0, stream>>>(Q, K, V, valid_lens, O);
}
